// TransferModel_85547158601842
// MI455X (gfx1250) — compile-verified, matrix operand order checked by reading, not run on hardware
//
#include <hip/hip_runtime.h>
#include <hip/hip_bf16.h>
#include <math.h>

typedef __attribute__((ext_vector_type(16))) __bf16 v16bf;
typedef __attribute__((ext_vector_type(8)))  float  v8f;

#define B_    8
#define TPRED 24
#define TTOT  25
#define N_    768
#define E_    24576
#define H_    64
#define M_    64
#define G3    192
#define ROWS  (B_*E_)   // 196608

// K index inside a 16-bit A/B fragment (ISA 7.12.2, 16-bit A 16x32 layout):
// lanes 0-15 half=0, lanes 16-31 half=1; VGPR v<4 -> K=2v+8*half+{0,1}; v>=4 -> K=16+2(v-4)+8*half+{0,1}
__device__ __forceinline__ int k_of(int i, int hf) {
    int v = i >> 1, lo = i & 1;
    int base = (v < 4) ? (2*v) : (16 + 2*(v-4));
    return base + 8*hf + lo;
}

// ---------------- setup kernels ----------------

__global__ void k_stats(const float* __restrict__ edge_attr, float* __restrict__ stats) {
    __shared__ double sh[256][4];
    int tid = threadIdx.x;
    double s0=0.0, s1=0.0, q0=0.0, q1=0.0;
    for (int e = tid; e < E_; e += 256) {
        double d = (double)edge_attr[2*e], r = (double)edge_attr[2*e+1];
        s0 += d; s1 += r; q0 += d*d; q1 += r*r;
    }
    sh[tid][0]=s0; sh[tid][1]=s1; sh[tid][2]=q0; sh[tid][3]=q1;
    __syncthreads();
    for (int off=128; off>0; off>>=1) {
        if (tid < off) { sh[tid][0]+=sh[tid+off][0]; sh[tid][1]+=sh[tid+off][1];
                         sh[tid][2]+=sh[tid+off][2]; sh[tid][3]+=sh[tid+off][3]; }
        __syncthreads();
    }
    if (tid==0) {
        double m0 = sh[0][0]/E_, m1 = sh[0][1]/E_;
        double v0 = (sh[0][2] - (double)E_*m0*m0)/(double)(E_-1);
        double v1 = (sh[0][3] - (double)E_*m1*m1)/(double)(E_-1);
        stats[0]=(float)m0; stats[1]=(float)m1;
        stats[2]=(float)sqrt(v0); stats[3]=(float)sqrt(v1);
    }
}

__global__ void k_prepack(const float* __restrict__ W_hh, const float* __restrict__ W1,
                          __bf16* __restrict__ whh_pk, __bf16* __restrict__ w1_pk) {
    int idx = blockIdx.x*blockDim.x + threadIdx.x;
    if (idx < 12*2*32*16) {     // W_hh: 12 N-tiles x 2 K-frags x 32 lanes x 16 elems
        int i = idx & 15; int lane = (idx>>4)&31; int kf = (idx>>9)&1; int t = idx>>10;
        int hf = lane>>4, nl = lane&15;
        int k = 32*kf + k_of(i, hf);
        whh_pk[idx] = (__bf16)W_hh[k*G3 + 16*t + nl];
    }
    if (idx < 4*2*32*16) {      // W1: 4 N-tiles x 2 K-frags
        int i = idx & 15; int lane = (idx>>4)&31; int kf = (idx>>9)&1; int t = idx>>10;
        int hf = lane>>4, nl = lane&15;
        int k = 32*kf + k_of(i, hf);
        w1_pk[idx] = (__bf16)W1[k*M_ + 16*t + nl];
    }
}

__global__ void k_hn(const float* __restrict__ pm25, const float* __restrict__ W_node,
                     const float* __restrict__ b_node, float* __restrict__ hn) {
    int p = blockIdx.x*blockDim.x + threadIdx.x;
    if (p < B_*N_) hn[p] = pm25[p]*W_node[0] + b_node[0];
}

// ---------------- per-step GRU + e_rep (WMMA) ----------------

#define STR 68   // padded LDS row stride (floats) -> conflict-free column reads

__global__ void __launch_bounds__(256)
k_gru(const float* __restrict__ feature, const int* __restrict__ ei,
      const float* __restrict__ edge_attr,
      const float* __restrict__ wind_mean, const float* __restrict__ wind_std,
      const float* __restrict__ W_ih, const float* __restrict__ b_ih,
      const float* __restrict__ b_hh,
      const float* __restrict__ b1, const float* __restrict__ W2,
      const float* __restrict__ b2, const float* __restrict__ stats,
      const __bf16* __restrict__ whh_pk, const __bf16* __restrict__ w1_pk,
      float* __restrict__ en, float* __restrict__ e_rep, int t_step)
{
    __shared__ float etile_s[8][16*STR];
    __shared__ float xbuf_s[8][16][3];
    const int wave = threadIdx.x >> 5;
    const int lane = threadIdx.x & 31;
    const int tile = blockIdx.x*8 + wave;
    const int p0 = tile*16;
    float* etile = etile_s[wave];
    float (*xbuf)[3] = xbuf_s[wave];
    const int hf = lane >> 4, nl = lane & 15;

    // ---- stage per-row edge-gate inputs (lanes 0..15, one row each) ----
    if (lane < 16) {
        int p = p0 + lane;
        int b = p / E_;
        int e = p - b*E_;
        int sn = ei[e];
        const float* f = feature + (((size_t)b*TTOT + (1+t_step))*N_ + sn)*4;
        float speed = f[2]*wind_std[0] + wind_mean[0];
        float wdir  = f[3]*wind_std[1] + wind_mean[1];
        float dist  = edge_attr[2*e];
        float direc = edge_attr[2*e+1];
        float theta = fabsf(direc - wdir);
        float ew = fmaxf(3.0f*speed*__cosf(theta)/dist, 0.0f);
        xbuf[lane][0] = (dist  - stats[0])/stats[2];
        xbuf[lane][1] = (direc - stats[1])/stats[3];
        xbuf[lane][2] = ew;
    }
    // ---- stage en tile (16x64, contiguous 1024 floats) into padded LDS ----
    {
        const float* esrc = en + (size_t)p0*H_;
        #pragma unroll
        for (int i=0;i<32;i++) {
            int idx = lane + 32*i;
            etile[(idx>>6)*STR + (idx&63)] = esrc[idx];
        }
    }
    __syncthreads();

    // ---- bf16 A-fragments of en_old (two 16x32 K-frags) ----
    v16bf a0, a1;
    #pragma unroll
    for (int i=0;i<16;i++) {
        a0[i] = (__bf16)etile[nl*STR +      k_of(i,hf)];
        a1[i] = (__bf16)etile[nl*STR + 32 + k_of(i,hf)];
    }

    // ---- gh = en_old @ W_hh : 12 C-tiles, K=64 -> 2 WMMAs each ----
    const v16bf* bw = (const v16bf*)whh_pk;
    v8f acc[12];
    #pragma unroll
    for (int t=0;t<12;t++) {
        v8f c = {};
        c = __builtin_amdgcn_wmma_f32_16x16x32_bf16(false, a0, false, bw[(t*2+0)*32+lane], (short)0, c, false, false);
        c = __builtin_amdgcn_wmma_f32_16x16x32_bf16(false, a1, false, bw[(t*2+1)*32+lane], (short)0, c, false, false);
        acc[t] = c;
    }

    // ---- gates + state update, in C-layout registers ----
    float* edst = en + (size_t)p0*H_;
    #pragma unroll
    for (int th=0; th<4; th++) {
        const int h  = 16*th + nl;
        const int hz = h + 64;
        const int hm = h + 128;
        float wr0=W_ih[h],  wr1=W_ih[G3+h],  wr2=W_ih[2*G3+h],  br=b_ih[h],  bhr=b_hh[h];
        float wz0=W_ih[hz], wz1=W_ih[G3+hz], wz2=W_ih[2*G3+hz], bz=b_ih[hz], bhz=b_hh[hz];
        float wn0=W_ih[hm], wn1=W_ih[G3+hm], wn2=W_ih[2*G3+hm], bn=b_ih[hm], bhn=b_hh[hm];
        #pragma unroll
        for (int s=0;s<8;s++) {
            int row = s + 8*hf;
            float x0=xbuf[row][0], x1=xbuf[row][1], x2=xbuf[row][2];
            float gr = x0*wr0 + x1*wr1 + x2*wr2 + br + acc[th  ][s] + bhr;
            float gz = x0*wz0 + x1*wz1 + x2*wz2 + bz + acc[th+4][s] + bhz;
            float gn = x0*wn0 + x1*wn1 + x2*wn2 + bn;
            float r = 1.0f/(1.0f + __expf(-gr));
            float z = 1.0f/(1.0f + __expf(-gz));
            float nn = tanhf(gn + r*(acc[th+8][s] + bhn));
            float eo = etile[row*STR + h];
            float enew = nn + z*(eo - nn);
            edst[row*H_ + h] = enew;       // persist state for next step
            etile[row*STR + h] = enew;     // reuse for A-fragments below
        }
    }
    __syncthreads();

    // ---- hid = relu(en_new @ W1 + b1);  e_rep = hid @ W2 + b2 ----
    #pragma unroll
    for (int i=0;i<16;i++) {
        a0[i] = (__bf16)etile[nl*STR +      k_of(i,hf)];
        a1[i] = (__bf16)etile[nl*STR + 32 + k_of(i,hf)];
    }
    const v16bf* bw1 = (const v16bf*)w1_pk;
    float part[8] = {0,0,0,0,0,0,0,0};
    #pragma unroll
    for (int t2=0;t2<4;t2++) {
        v8f c = {};
        c = __builtin_amdgcn_wmma_f32_16x16x32_bf16(false, a0, false, bw1[(t2*2+0)*32+lane], (short)0, c, false, false);
        c = __builtin_amdgcn_wmma_f32_16x16x32_bf16(false, a1, false, bw1[(t2*2+1)*32+lane], (short)0, c, false, false);
        float bb  = b1[16*t2+nl];
        float w2v = W2[16*t2+nl];
        #pragma unroll
        for (int s=0;s<8;s++)
            part[s] += fmaxf(c[s] + bb, 0.0f) * w2v;
    }
    #pragma unroll
    for (int s=0;s<8;s++) {
        float v = part[s];
        #pragma unroll
        for (int off=1; off<16; off<<=1) v += __shfl_xor(v, off, 32);
        if (nl == 0) e_rep[p0 + s + 8*hf] = v + b2[0];
    }
}

// ---------------- scatter + softmax/pred ----------------

__global__ void k_scatter(const int* __restrict__ ei, const float* __restrict__ e_rep,
                          float* __restrict__ Rout, int t_step) {
    int p = blockIdx.x*blockDim.x + threadIdx.x;
    if (p >= ROWS) return;
    int b = p / E_; int e = p - b*E_;
    int sn = ei[e], tn = ei[E_+e];
    size_t off = (((size_t)b*TPRED + t_step)*N_ + tn)*N_ + sn;
    Rout[off] = e_rep[p];   // dup indices: racy "one wins" == JAX scatter-set semantics
}

__global__ void __launch_bounds__(256)
k_softmax(float* __restrict__ Rout, const float* __restrict__ hn,
          float* __restrict__ pred, int t_step) {
    __shared__ float red[256];
    int rowid = blockIdx.x;              // 0..B*N-1
    int b = rowid / N_; int i = rowid - b*N_;
    float* row = Rout + (((size_t)b*TPRED + t_step)*N_ + i)*N_;
    int tid = threadIdx.x;
    float v[3], ex[3];
    float mx = -1e30f;
    #pragma unroll
    for (int k=0;k<3;k++) { v[k] = row[tid + 256*k]; mx = fmaxf(mx, v[k]); }
    red[tid] = mx; __syncthreads();
    for (int off=128;off>0;off>>=1){ if(tid<off) red[tid]=fmaxf(red[tid],red[tid+off]); __syncthreads(); }
    mx = red[0]; __syncthreads();
    float s = 0.0f;
    #pragma unroll
    for (int k=0;k<3;k++) { ex[k] = __expf(v[k]-mx); s += ex[k]; }
    red[tid] = s; __syncthreads();
    for (int off=128;off>0;off>>=1){ if(tid<off) red[tid]+=red[tid+off]; __syncthreads(); }
    s = red[0]; __syncthreads();
    float inv = 1.0f/s;
    float dot = 0.0f;
    const float* hb = hn + b*N_;
    #pragma unroll
    for (int k=0;k<3;k++) {
        float r = ex[k]*inv;
        row[tid + 256*k] = r;
        dot += r*hb[tid + 256*k];
    }
    red[tid] = dot; __syncthreads();
    for (int off=128;off>0;off>>=1){ if(tid<off) red[tid]+=red[tid+off]; __syncthreads(); }
    if (tid==0) pred[((size_t)b*TPRED + t_step)*N_ + i] = red[0];
}

// ---------------- launch ----------------

extern "C" void kernel_launch(void* const* d_in, const int* in_sizes, int n_in,
                              void* d_out, int out_size, void* d_ws, size_t ws_size,
                              hipStream_t stream) {
    const float* pm25      = (const float*)d_in[0];
    const float* feature   = (const float*)d_in[1];
    const int*   ei        = (const int*)  d_in[2];
    const float* edge_attr = (const float*)d_in[3];
    const float* wind_mean = (const float*)d_in[4];
    const float* wind_std  = (const float*)d_in[5];
    const float* W_ih      = (const float*)d_in[6];
    const float* W_hh      = (const float*)d_in[7];
    const float* b_ih      = (const float*)d_in[8];
    const float* b_hh      = (const float*)d_in[9];
    const float* W1        = (const float*)d_in[10];
    const float* b1        = (const float*)d_in[11];
    const float* W2        = (const float*)d_in[12];
    const float* b2        = (const float*)d_in[13];
    const float* W_node    = (const float*)d_in[14];
    const float* b_node    = (const float*)d_in[15];

    float* pred = (float*)d_out;
    float* Rout = pred + (size_t)B_*TPRED*N_;

    // workspace layout (256B-aligned slots)
    char* ws = (char*)d_ws;
    size_t off = 0;
    auto alloc = [&](size_t bytes) -> void* {
        void* p = ws + off;
        off += (bytes + 255) & ~(size_t)255;
        return p;
    };
    float*  stats  = (float*) alloc(4*sizeof(float));
    float*  hn     = (float*) alloc((size_t)B_*N_*sizeof(float));
    float*  e_rep  = (float*) alloc((size_t)ROWS*sizeof(float));
    __bf16* whh_pk = (__bf16*)alloc((size_t)12*2*32*16*sizeof(__bf16));
    __bf16* w1_pk  = (__bf16*)alloc((size_t)4*2*32*16*sizeof(__bf16));
    float*  en     = (float*) alloc((size_t)ROWS*H_*sizeof(float));

    hipMemsetAsync(en, 0, (size_t)ROWS*H_*sizeof(float), stream);                 // e0 = 0
    hipMemsetAsync(Rout, 0, (size_t)B_*TPRED*N_*N_*sizeof(float), stream);        // scatter baseline

    k_stats  <<<1, 256, 0, stream>>>(edge_attr, stats);
    k_prepack<<<48, 256, 0, stream>>>(W_hh, W1, whh_pk, w1_pk);
    k_hn     <<<(B_*N_+255)/256, 256, 0, stream>>>(pm25, W_node, b_node, hn);

    for (int t = 0; t < TPRED; t++) {
        k_gru    <<<ROWS/16/8, 256, 0, stream>>>(feature, ei, edge_attr, wind_mean, wind_std,
                                                 W_ih, b_ih, b_hh, b1, W2, b2, stats,
                                                 whh_pk, w1_pk, en, e_rep, t);
        k_scatter<<<(ROWS+255)/256, 256, 0, stream>>>(ei, e_rep, Rout, t);
        k_softmax<<<B_*N_, 256, 0, stream>>>(Rout, hn, pred, t);
    }
}
